// VariableSelectionNetwork_70136815944304
// MI455X (gfx1250) — compile-verified, matrix operand order checked by reading, not run on hardware
//
#include <hip/hip_runtime.h>
#include <hip/hip_bf16.h>

// ---------------------------------------------------------------------------
// Fused Variable Selection Network for MI455X (gfx1250, wave32, WMMA).
// One workgroup (256 thr = 8 waves) owns a 16-row tile; the whole (16 x 2048)
// "stacked" activation lives in LDS (64KB bf16) so no intermediate ever
// touches HBM.  All GEMMs use v_wmma_f32_16x16x32_bf16 with fp32 accumulate.
// Big weight matrices are pre-converted fp32->bf16 once into d_ws.
// ---------------------------------------------------------------------------

typedef __attribute__((ext_vector_type(16))) __bf16 v16bf;
typedef __attribute__((ext_vector_type(8)))  __bf16 v8bf;
typedef __attribute__((ext_vector_type(8)))  float  v8f;

#define N_TOTAL 16384   // B*T = 32*512
#define V  16
#define D  128
#define VD 2048
#define LN_EPS 1e-5f

__device__ __forceinline__ float eluf(float x) { return x > 0.f ? x : (expf(x) - 1.f); }
__device__ __forceinline__ float sigmf(float x) { return 1.f / (1.f + expf(-x)); }

__device__ __forceinline__ v8f wmma_bf16(v16bf a, v16bf b, v8f c) {
    // D = A(16x32) * B(32x16) + C, fp32 accumulate
    return __builtin_amdgcn_wmma_f32_16x16x32_bf16(
        /*neg_a=*/false, a, /*neg_b=*/false, b,
        /*c_mod=*/(short)0, c, /*reuse_a=*/false, /*reuse_b=*/false);
}

// reductions across the 16-lane half-wave (lane groups 0-15 / 16-31 stay separate)
__device__ __forceinline__ float hsum16(float v) {
    #pragma unroll
    for (int m = 1; m < 16; m <<= 1) v += __shfl_xor(v, m, 32);
    return v;
}
__device__ __forceinline__ float hmax16(float v) {
    #pragma unroll
    for (int m = 1; m < 16; m <<= 1) v = fmaxf(v, __shfl_xor(v, m, 32));
    return v;
}

// ---------------------------------------------------------------------------
// Prep: fp32 -> bf16 weight conversion into workspace
// ---------------------------------------------------------------------------
__global__ void cvt_bf16_kernel(const float* __restrict__ s, __bf16* __restrict__ d, int n) {
    int i = blockIdx.x * blockDim.x + threadIdx.x;
    if (i < n) d[i] = (__bf16)s[i];
}

// ---------------------------------------------------------------------------
// Main fused kernel: 1024 blocks x 256 threads, 16 rows per block.
// ---------------------------------------------------------------------------
__global__ __launch_bounds__(256)
void vsn_fused_kernel(const float* __restrict__ x,
                      const float* __restrict__ W1,  const float* __restrict__ b1,
                      const float* __restrict__ b2,  const float* __restrict__ bg,
                      const float* __restrict__ Ws,  const float* __restrict__ bs,
                      const float* __restrict__ gamma, const float* __restrict__ beta,
                      const float* __restrict__ sb1, const float* __restrict__ sb2,
                      const float* __restrict__ sbg, const float* __restrict__ sbs,
                      const float* __restrict__ sgamma, const float* __restrict__ sbeta,
                      const __bf16* __restrict__ w2bf,  const __bf16* __restrict__ wgbf,
                      const __bf16* __restrict__ sW1bf, const __bf16* __restrict__ sWsbf,
                      const __bf16* __restrict__ sW2bf, const __bf16* __restrict__ sWgbf,
                      float* __restrict__ out_proc, float* __restrict__ out_w)
{
    extern __shared__ char smem[];
    __bf16* flatL = (__bf16*)smem;                              // 16 x 2048 bf16 (64KB)
    __bf16* hsL   = (__bf16*)(smem + 65536);                    // 16 x 128 bf16 (4KB)
    float*  partL = (float*)(smem + 65536 + 4096);              // 8 waves x 32 x 8 f32 (8KB)
    float*  xsL   = (float*)(smem + 65536 + 4096 + 8192);       // 16 x 16 f32 (1KB)
    float*  wL    = (float*)(smem + 65536 + 4096 + 8192 + 1024);// 16 x 16 f32 (1KB)

    const int tid  = threadIdx.x;
    const int wv   = tid >> 5;        // wave id 0..7
    const int l    = tid & 31;        // lane
    const int n16  = l & 15;          // column index within a 16-wide tile
    const int half = l >> 4;          // K-half / row-half selector
    const int n0   = blockIdx.x * 16; // first row of this tile

    // x tile (16 rows x 16 vars) is contiguous in memory
    xsL[tid] = x[(size_t)n0 * V + tid];
    __syncthreads();

    const v8f vzero = {};

    // ===================== Phase A: 16 per-variable GRNs =====================
    // wave wv handles variables {wv, wv+8}
    #pragma unroll
    for (int vi = 0; vi < 2; ++vi) {
        const int v = wv + vi * 8;

        // Build A fragments in registers: A[m,k] = elu(x[m,v]*W1[v,k] + b1[v,k])
        // A layout (16-bit, 16x32): lane m=l%16 holds K = c*32 + half*16 + 0..15
        v16bf a[4];
        const float xv = xsL[n16 * V + v];
        #pragma unroll
        for (int c = 0; c < 4; ++c) {
            const int kb = c * 32 + half * 16;
            const float* w1p = W1 + v * D + kb;
            const float* b1p = b1 + v * D + kb;
            v16bf av;
            #pragma unroll
            for (int j = 0; j < 16; ++j)
                av[j] = (__bf16)eluf(xv * w1p[j] + b1p[j]);
            a[c] = av;
        }

        // 16x128 output = 8 N-tiles; keep both accumulator sets live for LN stats
        v8f accH[8], accG[8];
        #pragma unroll
        for (int t = 0; t < 8; ++t) { accH[t] = vzero; accG[t] = vzero; }

        #pragma unroll
        for (int t = 0; t < 8; ++t) {
            #pragma unroll
            for (int c = 0; c < 4; ++c) {
                // B[k,o] = W2[v][o][k]: lane o = t*16+n16, contiguous k
                const int off = v * D * D + (t * 16 + n16) * D + c * 32 + half * 16;
                v16bf bh = *(const v16bf*)(w2bf + off);
                v16bf bg_ = *(const v16bf*)(wgbf + off);
                accH[t] = wmma_bf16(a[c], bh, accH[t]);
                accG[t] = wmma_bf16(a[c], bg_, accG[t]);
            }
        }

        // Gate + skip (C layout: lane n=l%16 is col o=t*16+n; reg j is row m=j+8*half)
        float gamv[8], betv[8];
        #pragma unroll
        for (int t = 0; t < 8; ++t) {
            const int o = t * 16 + n16;
            const float b2o = b2[v * D + o], bgo = bg[v * D + o];
            const float Wso = Ws[v * D + o], bso = bs[v * D + o];
            gamv[t] = gamma[v * D + o];
            betv[t] = beta[v * D + o];
            #pragma unroll
            for (int j = 0; j < 8; ++j) {
                const int m = j + half * 8;
                const float xm   = xsL[m * V + v];
                const float hval = accH[t][j] + b2o;
                const float gval = sigmf(accG[t][j] + bgo);
                const float skip = xm * Wso + bso;
                accH[t][j] = gval * hval + (1.f - gval) * skip;   // y
            }
        }

        // LayerNorm over o (128 values per row): in-register + 16-lane reduce
        #pragma unroll
        for (int j = 0; j < 8; ++j) {
            float s = 0.f, q = 0.f;
            #pragma unroll
            for (int t = 0; t < 8; ++t) { const float y = accH[t][j]; s += y; q += y * y; }
            s = hsum16(s); q = hsum16(q);
            const float mean = s * (1.f / 128.f);
            const float var  = q * (1.f / 128.f) - mean * mean;
            const float rstd = rsqrtf(var + LN_EPS);
            const int m = j + half * 8;
            #pragma unroll
            for (int t = 0; t < 8; ++t) {
                const float yn = (accH[t][j] - mean) * rstd * gamv[t] + betv[t];
                flatL[m * VD + v * D + t * 16 + n16] = (__bf16)yn;
            }
        }
    }
    __syncthreads();

    // ===================== Phase B1: hs = elu(flat @ sW1^T + sb1) ============
    // wave wv computes output columns [wv*16, wv*16+16)
    {
        v8f acc = vzero;
        for (int kc = 0; kc < 64; ++kc) {
            v16bf af = *(const v16bf*)(flatL + n16 * VD + kc * 32 + half * 16);
            v16bf bf = *(const v16bf*)(sW1bf + (size_t)(wv * 16 + n16) * VD + kc * 32 + half * 16);
            acc = wmma_bf16(af, bf, acc);
        }
        const int o = wv * 16 + n16;
        const float sb1o = sb1[o];
        #pragma unroll
        for (int j = 0; j < 8; ++j) {
            const int m = j + half * 8;
            hsL[m * D + o] = (__bf16)eluf(acc[j] + sb1o);
        }
    }
    __syncthreads();

    // ===================== Phase B2: sks = flat @ sWs^T (split-K, 8 waves) ===
    {
        v8f acc = vzero;
        #pragma unroll
        for (int kk = 0; kk < 8; ++kk) {
            const int kc = wv * 8 + kk;
            v16bf af = *(const v16bf*)(flatL + n16 * VD + kc * 32 + half * 16);
            v16bf bf = *(const v16bf*)(sWsbf + n16 * VD + kc * 32 + half * 16);
            acc = wmma_bf16(af, bf, acc);
        }
        #pragma unroll
        for (int j = 0; j < 8; ++j) partL[wv * 256 + l * 8 + j] = acc[j];
    }
    __syncthreads();

    // ============ Phase B3: h2s/gs GEMMs + LN + softmax (wave 0) =============
    if (wv == 0) {
        v8f accH = vzero, accG = vzero;
        #pragma unroll
        for (int kc = 0; kc < 4; ++kc) {
            v16bf af = *(const v16bf*)(hsL + n16 * D + kc * 32 + half * 16);
            v16bf b2f = *(const v16bf*)(sW2bf + n16 * D + kc * 32 + half * 16);
            v16bf bgf = *(const v16bf*)(sWgbf + n16 * D + kc * 32 + half * 16);
            accH = wmma_bf16(af, b2f, accH);
            accG = wmma_bf16(af, bgf, accG);
        }
        const float sb2n = sb2[n16], sbgn = sbg[n16], sbsn = sbs[n16];
        const float sgn = sgamma[n16], sbn = sbeta[n16];
        float z[8];
        #pragma unroll
        for (int j = 0; j < 8; ++j) {
            float sk = sbsn;   // deterministic 8-way split-K reduction
            #pragma unroll
            for (int w = 0; w < 8; ++w) sk += partL[w * 256 + l * 8 + j];
            const float h2 = accH[j] + sb2n;
            const float g  = sigmf(accG[j] + sbgn);
            z[j] = g * h2 + (1.f - g) * sk;
        }
        #pragma unroll
        for (int j = 0; j < 8; ++j) {
            const float mean = hsum16(z[j]) * (1.f / 16.f);
            const float msq  = hsum16(z[j] * z[j]) * (1.f / 16.f);
            const float rstd = rsqrtf(msq - mean * mean + LN_EPS);
            const float zn   = (z[j] - mean) * rstd * sgn + sbn;
            const float mx = hmax16(zn);
            const float e  = expf(zn - mx);
            const float se = hsum16(e);
            const float wt = e / se;
            const int m = j + half * 8;
            out_w[(size_t)(n0 + m) * V + n16] = wt;
            wL[m * V + n16] = wt;
        }
    }
    __syncthreads();

    // ===================== Phase C: processed = sum_v stacked * weight =======
    {
        const int m  = tid >> 4;          // row 0..15
        const int c8 = (tid & 15) * 8;    // 8-wide d slice
        float acc[8];
        #pragma unroll
        for (int dd = 0; dd < 8; ++dd) acc[dd] = 0.f;
        #pragma unroll
        for (int v = 0; v < V; ++v) {
            const float wt = wL[m * V + v];
            v8bf dv = *(const v8bf*)(flatL + m * VD + v * D + c8);
            #pragma unroll
            for (int dd = 0; dd < 8; ++dd) acc[dd] += (float)dv[dd] * wt;
        }
        float* op = out_proc + (size_t)(n0 + m) * D + c8;
        #pragma unroll
        for (int dd = 0; dd < 8; ++dd) op[dd] = acc[dd];
    }
}

// ---------------------------------------------------------------------------
extern "C" void kernel_launch(void* const* d_in, const int* in_sizes, int n_in,
                              void* d_out, int out_size, void* d_ws, size_t ws_size,
                              hipStream_t stream) {
    const float* x      = (const float*)d_in[0];
    const float* W1     = (const float*)d_in[1];
    const float* b1     = (const float*)d_in[2];
    const float* W2     = (const float*)d_in[3];
    const float* b2     = (const float*)d_in[4];
    const float* Wg     = (const float*)d_in[5];
    const float* bg     = (const float*)d_in[6];
    const float* Ws     = (const float*)d_in[7];
    const float* bs     = (const float*)d_in[8];
    const float* gamma  = (const float*)d_in[9];
    const float* beta   = (const float*)d_in[10];
    const float* sW1    = (const float*)d_in[11];
    const float* sb1    = (const float*)d_in[12];
    const float* sW2    = (const float*)d_in[13];
    const float* sb2    = (const float*)d_in[14];
    const float* sWg    = (const float*)d_in[15];
    const float* sbg    = (const float*)d_in[16];
    const float* sWs    = (const float*)d_in[17];
    const float* sbs    = (const float*)d_in[18];
    const float* sgamma = (const float*)d_in[19];
    const float* sbeta  = (const float*)d_in[20];

    // bf16 weight pool in workspace (~1.6 MB)
    __bf16* ws    = (__bf16*)d_ws;
    __bf16* w2bf  = ws;                 // V*D*D = 262144
    __bf16* wgbf  = ws + 262144;        // 262144
    __bf16* sW1bf = ws + 524288;        // D*VD = 262144
    __bf16* sWsbf = ws + 786432;        // V*VD = 32768
    __bf16* sW2bf = ws + 819200;        // V*D  = 2048
    __bf16* sWgbf = ws + 821248;        // V*D  = 2048

    cvt_bf16_kernel<<<(262144 + 255) / 256, 256, 0, stream>>>(W2,  w2bf,  262144);
    cvt_bf16_kernel<<<(262144 + 255) / 256, 256, 0, stream>>>(Wg,  wgbf,  262144);
    cvt_bf16_kernel<<<(262144 + 255) / 256, 256, 0, stream>>>(sW1, sW1bf, 262144);
    cvt_bf16_kernel<<<(32768  + 255) / 256, 256, 0, stream>>>(sWs, sWsbf, 32768);
    cvt_bf16_kernel<<<(2048   + 255) / 256, 256, 0, stream>>>(sW2, sW2bf, 2048);
    cvt_bf16_kernel<<<(2048   + 255) / 256, 256, 0, stream>>>(sWg, sWgbf, 2048);

    float* out_proc = (float*)d_out;                        // (N, 128)
    float* out_w    = out_proc + (size_t)N_TOTAL * D;       // (N, 16)

    const size_t shmem = 65536 + 4096 + 8192 + 1024 + 1024; // 79872 B < 320 KB/WGP
    vsn_fused_kernel<<<N_TOTAL / 16, 256, shmem, stream>>>(
        x, W1, b1, b2, bg, Ws, bs, gamma, beta,
        sb1, sb2, sbg, sbs, sgamma, sbeta,
        w2bf, wgbf, sW1bf, sWsbf, sW2bf, sWgbf,
        out_proc, out_w);
}
